// Transformer_40389872451995
// MI455X (gfx1250) — matrix operand order repaired and checked by reading, not compiled, not run on hardware
//
#include <hip/hip_runtime.h>

#define DEPTH 6
#define DIM   512
#define HEADS 8
#define DHEAD 64
#define MLP_  2048
#define SEQ   760
#define SEQP  768
#define BATCH 16
#define INNER 512
#define NTOK  (BATCH*SEQ)   // 12160, divisible by 128

typedef __bf16 bf16;
typedef __attribute__((ext_vector_type(16))) __bf16 v16bf;
typedef __attribute__((ext_vector_type(8)))  float  v8f;
typedef __attribute__((ext_vector_type(4)))  int    v4i;

// ---- CDNA5 async global->LDS copy (ASYNCcnt) with safe fallback ----------
#if defined(__has_builtin)
# if __has_builtin(__builtin_amdgcn_global_load_async_to_lds_b128)
#  define HAS_ASYNC 1
# endif
# if __has_builtin(__builtin_amdgcn_s_wait_asynccnt)
#  define HAS_WAITA 1
# endif
#endif
#ifndef HAS_ASYNC
# define HAS_ASYNC 0
#endif
#ifndef HAS_WAITA
# define HAS_WAITA 0
#endif

#define AS1P __attribute__((address_space(1)))
#define AS3P __attribute__((address_space(3)))

__device__ inline void async_copy16(void* l, const void* g) {
#if HAS_ASYNC
  __builtin_amdgcn_global_load_async_to_lds_b128(
      (AS1P v4i*)(unsigned long long)g,
      (AS3P v4i*)(unsigned)(unsigned long long)l,
      0, 0);
#else
  *(uint4*)l = *(const uint4*)g;
#endif
}

__device__ inline void async_wait() {
#if HAS_ASYNC
# if HAS_WAITA
  __builtin_amdgcn_s_wait_asynccnt(0);
# else
  asm volatile("s_wait_asynccnt 0" ::: "memory");
# endif
#endif
}

// --------------------------------------------------------------------------
union FragU { v16bf v; uint4 q[2]; };

__device__ inline v8f zero8() {
  v8f z = {0.f,0.f,0.f,0.f,0.f,0.f,0.f,0.f};
  return z;
}

__device__ inline v8f wmma_bf16(v16bf a, v16bf b, v8f c) {
  return __builtin_amdgcn_wmma_f32_16x16x32_bf16(false, a, false, b, (short)0, c, false, false);
}

// A fragment: 16(M) x 32(K) bf16 from row-major source, ld in elements.
__device__ inline v16bf load_a_frag(const bf16* p, int ld) {
  int l = threadIdx.x & 31;
  const bf16* row = p + (size_t)(l & 15) * ld + ((l >> 4) << 3);
  FragU f;
  f.q[0] = *(const uint4*)(row);
  f.q[1] = *(const uint4*)(row + 16);
  return f.v;
}

// B fragment: 32(K) x 16(N) bf16 from N-major ([N][K]) source, ld in elements.
__device__ inline v16bf load_b_frag(const bf16* p, int ld) {
  int l = threadIdx.x & 31;
  const bf16* row = p + (size_t)(l & 15) * ld + ((l >> 4) << 3);
  FragU f;
  f.q[0] = *(const uint4*)(row);
  f.q[1] = *(const uint4*)(row + 16);
  return f.v;
}

// fp32 [d][K][N] -> bf16 [d][N][K]
__global__ void k_cvt_transpose(const float* __restrict__ src, bf16* __restrict__ dst,
                                int K, int N, int total) {
  int idx = blockIdx.x * 256 + threadIdx.x;
  if (idx >= total) return;
  int per = K * N;
  int d = idx / per, r = idx - d * per;
  int k = r / N, n = r - k * N;
  dst[(size_t)d * per + (size_t)n * K + k] = (bf16)src[idx];
}

// Additive key mask: 0 (valid) or -1e30 (pad / voxel==0), layer-invariant.
__global__ void k_mask(const int* __restrict__ voxel, float* __restrict__ maskf) {
  int idx = blockIdx.x * 256 + threadIdx.x;
  if (idx >= BATCH*SEQP) return;
  int b = idx / SEQP, j = idx - b*SEQP;
  float m = -1.0e30f;
  if (j < SEQ) {
    if (j == 0) m = 0.f;
    else m = (voxel[b*(SEQ-1) + j - 1] != 0) ? 0.f : -1.0e30f;
  }
  maskf[idx] = m;
}

// One wave per token: fp32 x[NTOK][512] -> bf16 LN output
__global__ void k_layernorm(const float* __restrict__ x, const float* __restrict__ g,
                            const float* __restrict__ beta, bf16* __restrict__ out) {
  int gw = (blockIdx.x * blockDim.x + threadIdx.x) >> 5;
  int lane = threadIdx.x & 31;
  if (gw >= NTOK) return;
  const float4* row = (const float4*)(x + (size_t)gw * DIM);
  float4 v[4];
  float s = 0.f, ss = 0.f;
#pragma unroll
  for (int i = 0; i < 4; i++) {
    v[i] = row[lane + 32*i];
    s  += v[i].x + v[i].y + v[i].z + v[i].w;
    ss += v[i].x*v[i].x + v[i].y*v[i].y + v[i].z*v[i].z + v[i].w*v[i].w;
  }
#pragma unroll
  for (int off = 1; off < 32; off <<= 1) {
    s  += __shfl_xor(s, off, 32);
    ss += __shfl_xor(ss, off, 32);
  }
  float mu  = s * (1.f/DIM);
  float var = ss * (1.f/DIM) - mu*mu;
  float rs  = rsqrtf(var + 1e-5f);
#pragma unroll
  for (int i = 0; i < 4; i++) {
    int j = lane + 32*i;
    float4 gv = ((const float4*)g)[j];
    float4 bv = ((const float4*)beta)[j];
    union { bf16 h[4]; uint2 u; } pk;
    pk.h[0] = (bf16)((v[i].x - mu)*rs*gv.x + bv.x);
    pk.h[1] = (bf16)((v[i].y - mu)*rs*gv.y + bv.y);
    pk.h[2] = (bf16)((v[i].z - mu)*rs*gv.z + bv.z);
    pk.h[3] = (bf16)((v[i].w - mu)*rs*gv.w + bv.w);
    *(uint2*)(out + (size_t)gw*DIM + 4*j) = pk.u;
  }
}

// C[M][N] = A[M][K](bf16) * Bt[N][K](bf16); 128x128 block tile, 8 waves,
// K-step 64, double-buffered async LDS staging.
// MODE 0: fp32 acc+bias+resid; MODE 1: bf16 acc (cols<INNER pre-scaled 1/8);
// MODE 2: bf16 gelu(acc+bias).
template<int MODE>
__global__ __launch_bounds__(256) void k_gemm(
    const bf16* __restrict__ A, const bf16* __restrict__ Bt,
    const float* __restrict__ bias, const float* __restrict__ resid,
    float* __restrict__ outf, bf16* __restrict__ outh,
    int M, int N, int K) {
  __shared__ bf16 smA[2][128*72];   // 128 rows x 64 k (+8 pad)
  __shared__ bf16 smB[2][128*72];
  int tid = threadIdx.x;
  int wid = tid >> 5, lane = tid & 31;
  int bm = blockIdx.y * 128, bn = blockIdx.x * 128;
  int wm = (wid >> 1) * 32, wn = (wid & 1) * 64;

  v8f acc[2][4];
#pragma unroll
  for (int i = 0; i < 2; i++)
#pragma unroll
    for (int j = 0; j < 4; j++) acc[i][j] = zero8();

  auto stage = [&](int buf, int k0) {
#pragma unroll
    for (int p = 0; p < 4; p++) {
      int c = p*256 + tid;                 // 1024 chunks of 8 bf16 per tile
      int r = c >> 3, cc = (c & 7) << 3;   // r in [0,128), cc in {0..56}
      async_copy16(&smA[buf][r*72 + cc], &A[(size_t)(bm + r)*K + (k0 + cc)]);
      async_copy16(&smB[buf][r*72 + cc], &Bt[(size_t)(bn + r)*K + (k0 + cc)]);
    }
  };

  stage(0, 0);
  async_wait();
  __syncthreads();
  int cur = 0;
  for (int k0 = 0; k0 < K; k0 += 64) {
    if (k0 + 64 < K) stage(cur ^ 1, k0 + 64);
    const bf16* sA = &smA[cur][0];
    const bf16* sB = &smB[cur][0];
#pragma unroll
    for (int ki = 0; ki < 2; ki++) {
      v16bf af[2], bfr[4];
#pragma unroll
      for (int i = 0; i < 2; i++) af[i] = load_a_frag(&sA[(wm + i*16)*72 + ki*32], 72);
#pragma unroll
      for (int j = 0; j < 4; j++) bfr[j] = load_b_frag(&sB[(wn + j*16)*72 + ki*32], 72);
#pragma unroll
      for (int i = 0; i < 2; i++)
#pragma unroll
        for (int j = 0; j < 4; j++)
          acc[i][j] = wmma_bf16(af[i], bfr[j], acc[i][j]);
    }
    async_wait();
    __syncthreads();
    cur ^= 1;
  }

  int col = lane & 15;
  int rb  = (lane >> 4) * 8;
#pragma unroll
  for (int i = 0; i < 2; i++) {
#pragma unroll
    for (int j = 0; j < 4; j++) {
      int gn = bn + wn + j*16 + col;
      float bv = (MODE == 1) ? 0.f : bias[gn];
      float scale = (MODE == 1 && gn < INNER) ? 0.125f : 1.f;  // q pre-scale
#pragma unroll
      for (int r = 0; r < 8; r++) {
        int gm = bm + wm + i*16 + rb + r;
        size_t idx = (size_t)gm * N + gn;
        float vv = acc[i][j][r] + bv;
        if (MODE == 0) {
          outf[idx] = vv + resid[idx];
        } else if (MODE == 1) {
          outh[idx] = (bf16)(vv * scale);
        } else { // exact GELU
          outh[idx] = (bf16)(0.5f * vv * (1.f + erff(vv * 0.70710678118654752f)));
        }
      }
    }
  }
}

// v section of qkv[tok][1536] -> vt[b][h][d][SEQP], zero-padded keys >= SEQ
__global__ void k_vtrans(const bf16* __restrict__ qkv, bf16* __restrict__ vt) {
  int idx = blockIdx.x * 256 + threadIdx.x;
  if (idx >= BATCH*HEADS*DHEAD*SEQP) return;
  int j = idx % SEQP; int t = idx / SEQP;
  int d = t % DHEAD;  t /= DHEAD;
  int h = t % HEADS;  int b = t / HEADS;
  bf16 val = (bf16)0.f;
  if (j < SEQ)
    val = qkv[(size_t)(b*SEQ + j)*(3*INNER) + 2*INNER + h*DHEAD + d];
  vt[idx] = val;
}

// Flash attention: one wave = 16 query rows; block cooperatively stages the
// shared 32-key K/V tiles (double-buffered async); all matmuls via WMMA.
__global__ __launch_bounds__(256) void k_attention(
    const bf16* __restrict__ qkv, const bf16* __restrict__ vt,
    const float* __restrict__ maskf, bf16* __restrict__ o) {
  __shared__ bf16 lK[2][32*72];   // 32 keys x 64 dims (+8 pad)
  __shared__ bf16 lV[2][64*40];   // 64 dims x 32 keys (+8 pad)
  __shared__ bf16 lP[8][16*32];   // per-wave P staging
  int tid = threadIdx.x, wid = tid >> 5, lane = tid & 31;
  const int nqb = SEQP / 128;     // 6
  int qb = blockIdx.x % nqb;
  int t  = blockIdx.x / nqb;
  int h  = t % HEADS, b = t / HEADS;
  int m0 = qb*128 + wid*16;

  const bf16* kgbase = qkv + (size_t)(b*SEQ)*(3*INNER) + INNER + h*DHEAD;
  const bf16* vgbase = vt + (size_t)((b*HEADS + h)*DHEAD)*SEQP;

  auto stage = [&](int buf, int n0) {
    {
      int r = tid >> 3, c = (tid & 7) << 3;     // 32 rows x 8 chunks
      async_copy16(&lK[buf][r*72 + c], kgbase + (size_t)(n0 + r)*(3*INNER) + c);
    }
    {
      int r = tid >> 2, c = (tid & 3) << 3;     // 64 rows x 4 chunks
      async_copy16(&lV[buf][r*40 + c], vgbase + (size_t)r*SEQP + (n0 + c));
    }
  };

  const bf16* qbase = qkv + (size_t)(b*SEQ + m0)*(3*INNER) + h*DHEAD;
  v16bf qf[2];
#pragma unroll
  for (int kc = 0; kc < 2; kc++) qf[kc] = load_a_frag(qbase + kc*32, 3*INNER);

  v8f oacc[4];
#pragma unroll
  for (int j = 0; j < 4; j++) oacc[j] = zero8();
  float mrun[8], lrun[8];
#pragma unroll
  for (int r = 0; r < 8; r++) { mrun[r] = -3.0e38f; lrun[r] = 0.f; }

  const int cj   = lane & 15;
  const int prow = (lane >> 4) * 8;
  bf16* P = &lP[wid][0];

  stage(0, 0);
  async_wait();
  __syncthreads();
  int cur = 0;
  for (int n0 = 0; n0 < SEQP; n0 += 32) {
    if (n0 + 32 < SEQP) stage(cur ^ 1, n0 + 32);
    v8f s0 = zero8(), s1 = zero8();
#pragma unroll
    for (int kc = 0; kc < 2; kc++) {
      v16bf kf0 = load_b_frag(&lK[cur][ 0*72 + kc*32], 72);
      v16bf kf1 = load_b_frag(&lK[cur][16*72 + kc*32], 72);
      s0 = wmma_bf16(qf[kc], kf0, s0);
      s1 = wmma_bf16(qf[kc], kf1, s1);
    }
    float mv0 = maskf[b*SEQP + n0 + cj];
    float mv1 = maskf[b*SEQP + n0 + 16 + cj];

    float p0[8], p1[8], bmax[8], rsum[8], alpha[8];
#pragma unroll
    for (int r = 0; r < 8; r++) {
      p0[r] = s0[r] + mv0;              // q pre-scaled by 1/8, mask additive
      p1[r] = s1[r] + mv1;
      bmax[r] = fmaxf(p0[r], p1[r]);
    }
#pragma unroll
    for (int off = 1; off < 16; off <<= 1)
#pragma unroll
      for (int r = 0; r < 8; r++)
        bmax[r] = fmaxf(bmax[r], __shfl_xor(bmax[r], off, 32));
#pragma unroll
    for (int r = 0; r < 8; r++) {
      float mn = fmaxf(mrun[r], bmax[r]);
      alpha[r] = __expf(mrun[r] - mn);
      mrun[r] = mn;
      p0[r] = __expf(p0[r] - mn);
      p1[r] = __expf(p1[r] - mn);
      rsum[r] = p0[r] + p1[r];
    }
#pragma unroll
    for (int off = 1; off < 16; off <<= 1)
#pragma unroll
      for (int r = 0; r < 8; r++)
        rsum[r] += __shfl_xor(rsum[r], off, 32);
#pragma unroll
    for (int r = 0; r < 8; r++) lrun[r] = lrun[r]*alpha[r] + rsum[r];
#pragma unroll
    for (int j = 0; j < 4; j++)
#pragma unroll
      for (int r = 0; r < 8; r++)
        oacc[j][r] *= alpha[r];

    // stage P (16x32, D-layout) into LDS, reload as A-fragment
#pragma unroll
    for (int r = 0; r < 8; r++) {
      P[(prow + r)*32 + cj]      = (bf16)p0[r];
      P[(prow + r)*32 + cj + 16] = (bf16)p1[r];
    }
    asm volatile("s_wait_dscnt 0" ::: "memory");
    v16bf pf = load_a_frag(P, 32);
#pragma unroll
    for (int dt = 0; dt < 4; dt++) {
      v16bf vf = load_b_frag(&lV[cur][(dt*16)*40], 40);
      oacc[dt] = wmma_bf16(pf, vf, oacc[dt]);
    }
    async_wait();
    __syncthreads();
    cur ^= 1;
  }

#pragma unroll
  for (int r = 0; r < 8; r++) lrun[r] = 1.f / lrun[r];
#pragma unroll
  for (int dt = 0; dt < 4; dt++)
#pragma unroll
    for (int r = 0; r < 8; r++) {
      int m = m0 + prow + r;
      if (m < SEQ)
        o[(size_t)(b*SEQ + m)*INNER + h*DHEAD + dt*16 + cj] = (bf16)(oacc[dt][r]*lrun[r]);
    }
}

static inline size_t alignup(size_t v) { return (v + 255) & ~(size_t)255; }

extern "C" void kernel_launch(void* const* d_in, const int* in_sizes, int n_in,
                              void* d_out, int out_size, void* d_ws, size_t ws_size,
                              hipStream_t stream) {
  (void)in_sizes; (void)n_in; (void)out_size; (void)ws_size;
  const float* in_x  = (const float*)d_in[0];
  const int*   voxel = (const int*)  d_in[1];
  // d_in[2] (cloud_len_list) unused by the reference body
  const float* ln1_g = (const float*)d_in[3];
  const float* ln1_b = (const float*)d_in[4];
  const float* w_qkv = (const float*)d_in[5];
  const float* w_out = (const float*)d_in[6];
  const float* b_out = (const float*)d_in[7];
  const float* ln2_g = (const float*)d_in[8];
  const float* ln2_b = (const float*)d_in[9];
  const float* w1    = (const float*)d_in[10];
  const float* b1    = (const float*)d_in[11];
  const float* w2    = (const float*)d_in[12];
  const float* b2    = (const float*)d_in[13];

  char* ws = (char*)d_ws;
  size_t off = 0;
  float* x   = (float*)(ws + off); off = alignup(off + (size_t)NTOK*DIM*4);
  bf16* tok  = (bf16*)(ws + off);  off = alignup(off + (size_t)NTOK*DIM*2);
  bf16* big  = (bf16*)(ws + off);  off = alignup(off + (size_t)NTOK*MLP_*2); // qkv & ffh share
  bf16* vt   = (bf16*)(ws + off);  off = alignup(off + (size_t)BATCH*HEADS*DHEAD*SEQP*2);
  float* mkf = (float*)(ws + off); off = alignup(off + (size_t)BATCH*SEQP*4);
  bf16* wqT  = (bf16*)(ws + off);  off = alignup(off + (size_t)DEPTH*3*INNER*DIM*2);
  bf16* woT  = (bf16*)(ws + off);  off = alignup(off + (size_t)DEPTH*DIM*INNER*2);
  bf16* w1T  = (bf16*)(ws + off);  off = alignup(off + (size_t)DEPTH*MLP_*DIM*2);
  bf16* w2T  = (bf16*)(ws + off);  off = alignup(off + (size_t)DEPTH*DIM*MLP_*2);

  (void)hipMemcpyAsync(x, in_x, (size_t)NTOK*DIM*4, hipMemcpyDeviceToDevice, stream);

  int tot;
  tot = DEPTH*DIM*(3*INNER);
  k_cvt_transpose<<<(tot+255)/256, 256, 0, stream>>>(w_qkv, wqT, DIM, 3*INNER, tot);
  tot = DEPTH*INNER*DIM;
  k_cvt_transpose<<<(tot+255)/256, 256, 0, stream>>>(w_out, woT, INNER, DIM, tot);
  tot = DEPTH*DIM*MLP_;
  k_cvt_transpose<<<(tot+255)/256, 256, 0, stream>>>(w1, w1T, DIM, MLP_, tot);
  tot = DEPTH*MLP_*DIM;
  k_cvt_transpose<<<(tot+255)/256, 256, 0, stream>>>(w2, w2T, MLP_, DIM, tot);
  tot = BATCH*SEQP;
  k_mask<<<(tot+255)/256, 256, 0, stream>>>(voxel, mkf);

  dim3 blk(256);
  for (int i = 0; i < DEPTH; i++) {
    k_layernorm<<<NTOK/8, blk, 0, stream>>>(x, ln1_g + i*DIM, ln1_b + i*DIM, tok);
    k_gemm<1><<<dim3(12, 95), blk, 0, stream>>>(tok, wqT + (size_t)i*3*INNER*DIM,
        nullptr, nullptr, nullptr, big, NTOK, 3*INNER, DIM);
    tot = BATCH*HEADS*DHEAD*SEQP;
    k_vtrans<<<(tot+255)/256, blk, 0, stream>>>(big, vt);
    k_attention<<<BATCH*HEADS*(SEQP/128), blk, 0, stream>>>(big, vt, mkf, tok);
    k_gemm<0><<<dim3(4, 95), blk, 0, stream>>>(tok, woT + (size_t)i*DIM*INNER,
        b_out + i*DIM, x, x, nullptr, NTOK, DIM, INNER);
    k_layernorm<<<NTOK/8, blk, 0, stream>>>(x, ln2_g + i*DIM, ln2_b + i*DIM, tok);
    k_gemm<2><<<dim3(16, 95), blk, 0, stream>>>(tok, w1T + (size_t)i*MLP_*DIM,
        b1 + i*MLP_, nullptr, nullptr, big, NTOK, MLP_, DIM);
    float* dst = (i == DEPTH-1) ? (float*)d_out : x;
    k_gemm<0><<<dim3(4, 95), blk, 0, stream>>>(big, w2T + (size_t)i*DIM*MLP_,
        b2 + i*DIM, x, dst, nullptr, NTOK, DIM, MLP_);
  }
}
